// CrossHeadMultiHeadAttention_8589934592314
// MI455X (gfx1250) — matrix operand order repaired and checked by reading, not compiled, not run on hardware
//
#include <hip/hip_runtime.h>
#include <hip/hip_bf16.h>

typedef __attribute__((ext_vector_type(16))) _Float16 v16h;
typedef __attribute__((ext_vector_type(8)))  float    v8f;
typedef __attribute__((ext_vector_type(4)))  float    v4f;

#define ROWS 128   // 16 tokens * 8 heads
#define SX   68    // f32 LDS row stride: 272B rows -> 16B aligned
#define SH   72    // f16 LDS row stride: 144B rows -> 16B aligned

// A-matrix (16x32 f16) fragment, contiguous k order (same as load_b_frag):
// lanes 0-15: M=lane,  half h -> K = h; lanes 16-31: K = 16 + h.
// Halves 0-15 are 32 contiguous bytes -> two ds_load_b128.
__device__ __forceinline__ v16h load_a_frag(const _Float16* A, int mbase, int kbase, int lane) {
  const int hi = (lane >> 4) & 1;
  const int m  = mbase + (lane & 15);
  v16h a;
#pragma unroll
  for (int h = 0; h < 16; ++h) {
    const int k = kbase + 16 * hi + h;
    a[h] = A[m * SH + k];
  }
  return a;
}

// B-matrix (32x16 f16) fragment: B[k][n] = W[n][k] (torch Linear: y = x W^T).
// lanes 0-15 hold K=kbase..+15, lanes 16-31 hold K=kbase+16..+31; N = lane&15.
__device__ __forceinline__ v16h load_b_frag(const float* __restrict__ W, int nbase, int kbase, int lane) {
  const int hi = (lane >> 4) & 1;
  const int n  = nbase + (lane & 15);
  v16h b;
#pragma unroll
  for (int h = 0; h < 16; ++h)
    b[h] = (_Float16)W[n * 64 + kbase + 16 * hi + h];
  return b;
}

// f16 pre-converted weights: 16 consecutive halfs = 32B -> two global_load_b128, no cvt.
__device__ __forceinline__ v16h load_b_frag(const _Float16* __restrict__ W, int nbase, int kbase, int lane) {
  const int hi = (lane >> 4) & 1;
  const int n  = nbase + (lane & 15);
  v16h b;
#pragma unroll
  for (int h = 0; h < 16; ++h)
    b[h] = W[n * 64 + kbase + 16 * hi + h];
  return b;
}

// One-shot weight conversion: Wq|Wk|Wv|Wo (4 x 64x64 f32) -> f16 in workspace.
__global__ __launch_bounds__(256) void convert_weights_kernel(
    const float* __restrict__ Wq, const float* __restrict__ Wk,
    const float* __restrict__ Wv, const float* __restrict__ Wo,
    _Float16* __restrict__ dst) {
  const int i = blockIdx.x * 256 + threadIdx.x;   // 0..4095
  dst[0 * 4096 + i] = (_Float16)Wq[i];
  dst[1 * 4096 + i] = (_Float16)Wk[i];
  dst[2 * 4096 + i] = (_Float16)Wv[i];
  dst[3 * 4096 + i] = (_Float16)Wo[i];
}

template <typename WT>
__global__ __launch_bounds__(256) void cross_head_attn_kernel(
    const float* __restrict__ x,
    const WT* __restrict__ Wq, const float* __restrict__ bq,
    const WT* __restrict__ Wk, const float* __restrict__ bk,
    const WT* __restrict__ Wv, const float* __restrict__ bv,
    const WT* __restrict__ Wo, const float* __restrict__ bo,
    const float* __restrict__ ln_g, const float* __restrict__ ln_b,
    float* __restrict__ out)
{
  __shared__ float    Xf[ROWS * SX];  // raw tile (residual), later final output
  __shared__ _Float16 Xn[ROWS * SH];  // LN'd activations, reused as attn output
  __shared__ _Float16 Qh[ROWS * SH];
  __shared__ _Float16 Kh[ROWS * SH];
  __shared__ _Float16 Vh[ROWS * SH];

  const int t   = threadIdx.x;
  const int blk = blockIdx.x;         // 4096 tiles
  const int bb_ = blk >> 8;           // batch index (256 tiles per batch)
  const int s0  = (blk & 255) * 16;   // first token of tile
  const size_t xbase = (size_t)bb_ * 512 * 4096 + (size_t)s0;
  const v4f* __restrict__ x4 = (const v4f*)x;

  // ---- coalesced wide NT load: x[b, nh, dd, s0..s0+15] -> Xf[token*8+nh][dd] ----
#pragma unroll
  for (int it = 0; it < 8; ++it) {
    const int q  = t + it * 256;      // quad index (2048 per tile)
    const int r  = q >> 2;            // feature row nh*64 + dd
    const int i0 = (q & 3) << 2;      // first local token of the quad
    const v4f v = __builtin_nontemporal_load(&x4[(xbase + (size_t)r * 4096 + i0) >> 2]);
    const int nh = r >> 6, dd = r & 63;
    Xf[((i0 + 0) * 8 + nh) * SX + dd] = v[0];
    Xf[((i0 + 1) * 8 + nh) * SX + dd] = v[1];
    Xf[((i0 + 2) * 8 + nh) * SX + dd] = v[2];
    Xf[((i0 + 3) * 8 + nh) * SX + dd] = v[3];
  }
  __syncthreads();

  // ---- LayerNorm over d=64 per (token, head) row ----
  if (t < ROWS) {
    float mu = 0.f;
#pragma unroll
    for (int dd = 0; dd < 64; ++dd) mu += Xf[t * SX + dd];
    mu *= (1.0f / 64.0f);
    float var = 0.f;
#pragma unroll
    for (int dd = 0; dd < 64; ++dd) { const float dv = Xf[t * SX + dd] - mu; var += dv * dv; }
    var *= (1.0f / 64.0f);
    const float rs = rsqrtf(var + 1e-5f);
#pragma unroll
    for (int dd = 0; dd < 64; ++dd) {
      const float xv = (Xf[t * SX + dd] - mu) * rs * ln_g[dd] + ln_b[dd];
      Xn[t * SH + dd] = (_Float16)xv;
    }
  }
  __syncthreads();

  const int wv   = t >> 5;   // wave id = M-tile (8 waves, 8 tiles of 16 rows)
  const int lane = t & 31;
  const int hi   = lane >> 4;
  const int lr   = lane & 15;

  // ---- Q, K, V projections: [128x64] @ [64x64]^T via WMMA f16 ----
  {
    const v16h a0 = load_a_frag(Xn, wv * 16, 0,  lane);
    const v16h a1 = load_a_frag(Xn, wv * 16, 32, lane);
    const WT*    Ws[3] = {Wq, Wk, Wv};
    const float* Bs[3] = {bq, bk, bv};
    _Float16*    Ds[3] = {Qh, Kh, Vh};
#pragma unroll
    for (int p = 0; p < 3; ++p) {
#pragma unroll
      for (int nt = 0; nt < 4; ++nt) {
        const v16h b0 = load_b_frag(Ws[p], nt * 16, 0,  lane);
        const v16h b1 = load_b_frag(Ws[p], nt * 16, 32, lane);
        v8f c = {};
        c = __builtin_amdgcn_wmma_f32_16x16x32_f16(false, a0, false, b0, (short)0, c, false, false);
        c = __builtin_amdgcn_wmma_f32_16x16x32_f16(false, a1, false, b1, (short)0, c, false, false);
        const float bias = Bs[p][nt * 16 + lr];
#pragma unroll
        for (int r = 0; r < 8; ++r)
          Ds[p][(wv * 16 + r + 8 * hi) * SH + nt * 16 + lr] = (_Float16)(c[r] + bias);
      }
    }
  }
  __syncthreads();

  // ---- attention across the 8 heads, per 16-wide sub-head (A=4), scale=1/sqrt(16) ----
  {
    const int tok = t >> 4;          // 0..15
    const int ah  = (t >> 2) & 3;    // sub-head 0..3
    const int qg  = t & 3;           // 2 query rows each
    const int rb  = tok * 8;
    const int cb  = ah * 16;         // 32B-aligned column base -> ds_load_b128 pairs
#pragma unroll
    for (int qq = 0; qq < 2; ++qq) {
      const int qi = qg * 2 + qq;
      float q[16];
#pragma unroll
      for (int d = 0; d < 16; ++d) q[d] = (float)Qh[(rb + qi) * SH + cb + d];
      float sc[8]; float mx = -3.0e38f;
#pragma unroll
      for (int j = 0; j < 8; ++j) {
        float s = 0.f;
#pragma unroll
        for (int d = 0; d < 16; ++d) s += q[d] * (float)Kh[(rb + j) * SH + cb + d];
        s *= 0.25f;                  // SUB^-0.5
        sc[j] = s;
        mx = fmaxf(mx, s);
      }
      float sum = 0.f;
#pragma unroll
      for (int j = 0; j < 8; ++j) { sc[j] = __expf(sc[j] - mx); sum += sc[j]; }
      const float inv = 1.0f / sum;
      float o[16];
#pragma unroll
      for (int d = 0; d < 16; ++d) o[d] = 0.f;
#pragma unroll
      for (int j = 0; j < 8; ++j) {
        const float w = sc[j] * inv;
#pragma unroll
        for (int d = 0; d < 16; ++d) o[d] += w * (float)Vh[(rb + j) * SH + cb + d];
      }
#pragma unroll
      for (int d = 0; d < 16; ++d)
        Xn[(rb + qi) * SH + cb + d] = (_Float16)o[d];   // reuse Xn as attention output
    }
  }
  __syncthreads();

  // ---- output projection + bias + residual (accumulate into Xf) ----
  {
    const v16h a0 = load_a_frag(Xn, wv * 16, 0,  lane);
    const v16h a1 = load_a_frag(Xn, wv * 16, 32, lane);
#pragma unroll
    for (int nt = 0; nt < 4; ++nt) {
      const v16h b0 = load_b_frag(Wo, nt * 16, 0,  lane);
      const v16h b1 = load_b_frag(Wo, nt * 16, 32, lane);
      v8f c = {};
      c = __builtin_amdgcn_wmma_f32_16x16x32_f16(false, a0, false, b0, (short)0, c, false, false);
      c = __builtin_amdgcn_wmma_f32_16x16x32_f16(false, a1, false, b1, (short)0, c, false, false);
      const float bias = bo[nt * 16 + lr];
#pragma unroll
      for (int r = 0; r < 8; ++r) {
        const int M   = wv * 16 + r + 8 * hi;
        const int col = nt * 16 + lr;
        Xf[M * SX + col] += c[r] + bias;     // residual already there
      }
    }
  }
  __syncthreads();

  // ---- coalesced wide NT writeout: Xf[token*8+nh][dd] -> out[b, nh, dd, s0..s0+15] ----
  v4f* __restrict__ out4 = (v4f*)out;
#pragma unroll
  for (int it = 0; it < 8; ++it) {
    const int q  = t + it * 256;
    const int r  = q >> 2;
    const int i0 = (q & 3) << 2;
    const int nh = r >> 6, dd = r & 63;
    v4f v;
    v[0] = Xf[((i0 + 0) * 8 + nh) * SX + dd];
    v[1] = Xf[((i0 + 1) * 8 + nh) * SX + dd];
    v[2] = Xf[((i0 + 2) * 8 + nh) * SX + dd];
    v[3] = Xf[((i0 + 3) * 8 + nh) * SX + dd];
    __builtin_nontemporal_store(v, &out4[(xbase + (size_t)r * 4096 + i0) >> 2]);
  }
}

extern "C" void kernel_launch(void* const* d_in, const int* in_sizes, int n_in,
                              void* d_out, int out_size, void* d_ws, size_t ws_size,
                              hipStream_t stream) {
  (void)in_sizes; (void)n_in; (void)out_size;
  const float* x   = (const float*)d_in[0];
  const float* Wq  = (const float*)d_in[1];
  const float* bq  = (const float*)d_in[2];
  const float* Wk  = (const float*)d_in[3];
  const float* bk  = (const float*)d_in[4];
  const float* Wv  = (const float*)d_in[5];
  const float* bv  = (const float*)d_in[6];
  const float* Wo  = (const float*)d_in[7];
  const float* bo  = (const float*)d_in[8];
  const float* g   = (const float*)d_in[9];
  const float* bb  = (const float*)d_in[10];
  float* out = (float*)d_out;

  if (ws_size >= (size_t)(4 * 4096 * sizeof(_Float16))) {
    // Convert weights to f16 once, then run the WMMA kernel on f16 weights.
    _Float16* wh = reinterpret_cast<_Float16*>(d_ws);
    convert_weights_kernel<<<dim3(16), dim3(256), 0, stream>>>(Wq, Wk, Wv, Wo, wh);
    cross_head_attn_kernel<_Float16><<<dim3(4096), dim3(256), 0, stream>>>(
        x, wh + 0 * 4096, bq, wh + 1 * 4096, bk, wh + 2 * 4096, bv, wh + 3 * 4096, bo,
        g, bb, out);
  } else {
    // Fallback: convert weights in-kernel (no workspace available).
    cross_head_attn_kernel<float><<<dim3(4096), dim3(256), 0, stream>>>(
        x, Wq, bq, Wk, bk, Wv, bv, Wo, bo, g, bb, out);
  }
}
